// MultiHeadSelfAttention_78443282694384
// MI455X (gfx1250) — matrix operand order repaired and checked by reading, not compiled, not run on hardware
//
#include <hip/hip_runtime.h>
#include <hip/hip_bf16.h>

// ---------------------------------------------------------------------------
// MI455X (gfx1250) multi-head causal self-attention.
// Compute-bound (~69 GFLOP vs ~120MB HBM traffic @ 23.3TB/s), so every matmul
// stage (QKV proj, QK^T, P@V, out proj) runs on v_wmma_f32_16x16x32_f16 with
// f32 accumulation. Data movement uses the CDNA5-specific engines where they
// fit: TDM (tensor_load_to_lds, TENSORcnt) for the padded GEMM A tile, and
// GLOBAL_LOAD_ASYNC_TO_LDS_B128 (ASYNCcnt) for the flash-attention K tile.
// wave32 everywhere.
// ---------------------------------------------------------------------------

typedef __attribute__((ext_vector_type(8)))  _Float16 v8h;
typedef __attribute__((ext_vector_type(16))) _Float16 v16h;
typedef __attribute__((ext_vector_type(8)))  float    v8f;
typedef __attribute__((ext_vector_type(4)))  unsigned int u32x4;
typedef __attribute__((ext_vector_type(8)))  unsigned int u32x8;
// Exact pointee type the async-LDS builtins expect (per clang diagnostic):
typedef __attribute__((__vector_size__(4 * sizeof(int)))) int i32x4_v;

#define BS   2
#define SEQ  2048
#define DIN  1024
#define HID  1024
#define NH   16
#define HD   64          // head dim
#define MX   (BS*SEQ)    // 4096 rows through the projections

// Feature gates for CDNA5 async paths (easy to flip off if codegen rejects).
#ifndef ATHENA_TRY_TDM
#define ATHENA_TRY_TDM 1
#endif
#if __has_builtin(__builtin_amdgcn_global_load_async_to_lds_b128)
#define ATHENA_HAVE_ASYNC 1
#endif

#define AS1CAST(p) ((__attribute__((address_space(1))) i32x4_v*)(p))
#define AS3CAST(p) ((__attribute__((address_space(3))) i32x4_v*)(p))

// -------------------- WMMA helpers (CDNA5 §7.12.2 layouts) ------------------

__device__ __forceinline__ v8f wmma_f16(v16h a, v16h b, v8f c) {
  // 8 args: (neg_a, A, neg_b, B, c_mod, C, reuse_a, reuse_b)
  return __builtin_amdgcn_wmma_f32_16x16x32_f16(false, a, false, b,
                                                (short)0, c, false, false);
}

// A-matrix 16x32 f16 fragment from a row-major tile base[row*ld + k].
// Lane (lo=lane&15, hi=lane>>4): row M = lo; halves 0..7 -> K = hi*8 + i,
// halves 8..15 -> K = 16 + hi*8 + i. ld multiple of 8 halves -> aligned b128.
__device__ __forceinline__ v16h ld_frag_a(const _Float16* base, int ld, int lane) {
  const int lo = lane & 15, hi = lane >> 4;
  const _Float16* p = base + (size_t)lo * ld + hi * 8;
  v8h x0 = *(const v8h*)(p);
  v8h x1 = *(const v8h*)(p + 16);
  v16h r;
#pragma unroll
  for (int i = 0; i < 8; ++i) { r[i] = x0[i]; r[i + 8] = x1[i]; }
  return r;
}

// B-matrix 32x16 f16 fragment from a TRANSPOSED tile base[n*ld + k] (B^T
// row-major). Lane: col N = lo; halves 0..7 -> K = hi*8 + i,
// halves 8..15 -> K = 16 + hi*8 + i (same order as ld_frag_a).
__device__ __forceinline__ v16h ld_frag_bt(const _Float16* base, int ld, int lane) {
  const int lo = lane & 15, hi = lane >> 4;
  const _Float16* p = base + (size_t)lo * ld + hi * 8;
  v8h x0 = *(const v8h*)(p);
  v8h x1 = *(const v8h*)(p + 16);
  v16h r;
#pragma unroll
  for (int i = 0; i < 8; ++i) { r[i] = x0[i]; r[i + 8] = x1[i]; }
  return r;
}

// -------------------- fp32 -> f16 conversion --------------------------------

__global__ void cvt_f32_f16_kernel(const float* __restrict__ s,
                                   _Float16* __restrict__ d, int n) {
  int i = blockIdx.x * blockDim.x + threadIdx.x;
  if (i < n) d[i] = (_Float16)s[i];
}

// -------------------- tiled GEMM + bias: C = A(f16) @ B(f16) + bias ---------
// Block tile 128x128, 256 threads = 8 waves arranged 4(M) x 2(N); each wave
// owns a 32x64 sub-tile = 2x4 WMMA C-fragments. K staged in 32-wide slices.
// A slice is staged by the Tensor Data Mover with hardware row padding.

#define BMT 128
#define BNT 128
#define BKT 32
#define LDA_S 40   // 32 + 8 halves pad (= TDM pad_amount 4 DWORDs per 16-DWORD row)
#define LDB_S 40

template <bool OUT_HALF>
__global__ __launch_bounds__(256) void gemm_bias_kernel(
    const _Float16* __restrict__ A,   // [M,K] row-major f16
    const _Float16* __restrict__ B,   // [K,N] row-major f16
    const float*    __restrict__ bias,// [N] f32
    void*           __restrict__ C,   // [M,N] f16 or f32
    int M, int N, int K) {
  __shared__ _Float16 sA[BMT * LDA_S];   // 10 KB
  __shared__ _Float16 sB[BNT * LDB_S];   // 10 KB (holds B^T slice)

  const int tid  = threadIdx.x;
  const int lane = tid & 31;
  const int wave = tid >> 5;
  const int wm   = wave & 3;     // 32-row strip
  const int wn   = wave >> 2;    // 64-col strip
  const int m0   = blockIdx.y * BMT;
  const int n0   = blockIdx.x * BNT;

#if ATHENA_TRY_TDM
  // 32-bit LDS byte offset of sA for the D# descriptor.
  const unsigned sA_lds =
      (unsigned)(unsigned long long)(__attribute__((address_space(3))) _Float16*)sA;
#endif

  v8f acc[2][4];
#pragma unroll
  for (int i = 0; i < 2; ++i)
#pragma unroll
    for (int j = 0; j < 4; ++j) acc[i][j] = (v8f)(0.0f);

  for (int k0 = 0; k0 < K; k0 += BKT) {
#if ATHENA_TRY_TDM
    // --- TDM: DMA the 128x32 A slice into LDS with pad 4dw per 16dw row ---
    if (wave == 0) {
      const unsigned long long ga =
          (unsigned long long)(const void*)(A + (size_t)m0 * K + k0);
      u32x4 g0;
      g0[0] = 1u;                                         // count=1, user mode
      g0[1] = sA_lds;                                     // lds_addr
      g0[2] = (unsigned)(ga & 0xffffffffu);               // global_addr[31:0]
      g0[3] = (unsigned)((ga >> 32) & 0x01ffffffu)        // global_addr[56:32]
              | (2u << 30);                               // type=2 ("image")
      u32x8 g1;
      g1[0] = (1u << 16)                                  // data_size = 2B
              | (1u << 20)                                // pad_enable
              | (3u << 22)                                // pad_interval: 16 DW
              | (3u << 25);                               // pad_amount:   4 DW
      g1[1] = ((unsigned)K & 0xffffu) << 16;              // tensor_dim0 lo16
      g1[2] = ((unsigned)K >> 16)                         // tensor_dim0 hi16
              | (((unsigned)M & 0xffffu) << 16);          // tensor_dim1 lo16
      g1[3] = ((unsigned)M >> 16) | ((unsigned)BKT << 16);// dim1 hi | tile_dim0
      g1[4] = (unsigned)BMT;                              // tile_dim1 (tile_dim2=0)
      g1[5] = (unsigned)K;                                // tensor_dim0_stride lo32
      g1[6] = 0u;                                         // stride0 hi | stride1 lo
      g1[7] = 0u;
      asm volatile("tensor_load_to_lds %0, %1" :: "s"(g0), "s"(g1) : "memory");
    }
#else
    // --- manual A staging: 128 rows x 32 halves, 16 halves per thread ---
    {
      const int r = tid >> 1, s = (tid & 1) * 16;
      const _Float16* g = A + (size_t)(m0 + r) * K + k0 + s;
      v8h u0 = *(const v8h*)g;
      v8h u1 = *(const v8h*)(g + 8);
      *(v8h*)&sA[r * LDA_S + s]     = u0;
      *(v8h*)&sA[r * LDA_S + s + 8] = u1;
    }
#endif
    // --- stage B slice transposed: sB[n*LDB + k] = B[(k0+k)*N + n0+n] ---
    {
      const int kb = tid >> 4;          // 0..15
      const int nb = (tid & 15) * 8;    // 0..120
#pragma unroll
      for (int kk = kb; kk < BKT; kk += 16) {
        const _Float16* g = B + (size_t)(k0 + kk) * N + n0 + nb;
        __builtin_prefetch(g + BKT * N, 0, 0);   // global_prefetch_b8 next slice
        v8h u = *(const v8h*)g;
#pragma unroll
        for (int j = 0; j < 8; ++j) sB[(nb + j) * LDB_S + kk] = u[j];
      }
    }
#if ATHENA_TRY_TDM
    __builtin_amdgcn_s_wait_tensorcnt(0);   // no-op for waves with TENSORcnt==0
#endif
    __syncthreads();

    v16h af[2], bf[4];
#pragma unroll
    for (int mi = 0; mi < 2; ++mi)
      af[mi] = ld_frag_a(&sA[(wm * 32 + mi * 16) * LDA_S], LDA_S, lane);
#pragma unroll
    for (int ni = 0; ni < 4; ++ni)
      bf[ni] = ld_frag_bt(&sB[(wn * 64 + ni * 16) * LDB_S], LDB_S, lane);
#pragma unroll
    for (int mi = 0; mi < 2; ++mi)
#pragma unroll
      for (int ni = 0; ni < 4; ++ni)
        acc[mi][ni] = wmma_f16(af[mi], bf[ni], acc[mi][ni]);
    __syncthreads();
  }

  // Epilogue. C-fragment: row = v + 8*hi, col = lo.
  const int lo = lane & 15, hi = lane >> 4;
#pragma unroll
  for (int mi = 0; mi < 2; ++mi)
#pragma unroll
    for (int ni = 0; ni < 4; ++ni) {
      const int n = n0 + wn * 64 + ni * 16 + lo;
      const float bv = bias[n];
#pragma unroll
      for (int v = 0; v < 8; ++v) {
        const int m = m0 + wm * 32 + mi * 16 + hi * 8 + v;
        const float val = acc[mi][ni][v] + bv;
        if (OUT_HALF) ((_Float16*)C)[(size_t)m * N + n] = (_Float16)val;
        else          ((float*)C)[(size_t)m * N + n]    = val;
      }
    }
}

// -------------------- flash attention (causal, online softmax) --------------
// One block per (b, h, 64-query tile). 128 threads = 4 waves; wave w owns
// 16 queries. K tile staged via GLOBAL_LOAD_ASYNC_TO_LDS_B128 (ASYNCcnt),
// V^T staged manually (transpose). Softmax row state per-lane: row = v+8*hi,
// so each lane tracks 8 rows; row reductions are 16-lane shfl_xor trees.

#define LDK_S 72   // 64 + 8 halves pad (144B rows: multiple of 16B)

__global__ __launch_bounds__(128) void flash_attn_kernel(
    const _Float16* __restrict__ Q,   // [B*S, H*D]
    const _Float16* __restrict__ Km,  // [B*S, H*D]
    const _Float16* __restrict__ Vm,  // [B*S, H*D]
    _Float16*       __restrict__ O) { // [B*S, H*D]
  __shared__ _Float16 sK [64 * LDK_S];       // [key][d]
  __shared__ _Float16 sVt[64 * LDK_S];       // [d][key]  (B^T for P@V)
  __shared__ _Float16 sP [4 * 16 * LDK_S];   // per-wave P tiles [16][keys]

  const int tid  = threadIdx.x;
  const int lane = tid & 31;
  const int wave = tid >> 5;
  const int lo   = lane & 15, hi = lane >> 4;

  const int b  = blockIdx.z;
  const int h  = blockIdx.y;
  const int q0 = blockIdx.x * 64;
  const size_t rstr = (size_t)NH * HD;   // 1024 halves per token row

  const _Float16* Qb = Q  + ((size_t)b * SEQ) * rstr + (size_t)h * HD;
  const _Float16* Kb = Km + ((size_t)b * SEQ) * rstr + (size_t)h * HD;
  const _Float16* Vb = Vm + ((size_t)b * SEQ) * rstr + (size_t)h * HD;

  // Q fragments for this wave's 16 rows, K-dim = d in {0..31, 32..63}.
  v16h aq[2];
#pragma unroll
  for (int kk = 0; kk < 2; ++kk)
    aq[kk] = ld_frag_a(Qb + (size_t)(q0 + wave * 16) * rstr + kk * 32,
                       (int)rstr, lane);

  v8f o[4];
#pragma unroll
  for (int ni = 0; ni < 4; ++ni) o[ni] = (v8f)(0.0f);
  float rm[8], rs[8];
#pragma unroll
  for (int v = 0; v < 8; ++v) { rm[v] = -1.0e30f; rs[v] = 0.0f; }

  for (int kt0 = 0; kt0 <= q0; kt0 += 64) {
    // --- stage K tile: contiguous rows -> async LDS copy engine ---
    {
      const int key = tid >> 1, seg = (tid & 1) * 32;
      const _Float16* g = Kb + (size_t)(kt0 + key) * rstr + seg;
      _Float16* l = &sK[key * LDK_S + seg];
#if defined(ATHENA_HAVE_ASYNC)
      __builtin_amdgcn_global_load_async_to_lds_b128(AS1CAST(g), AS3CAST(l), 0, 0);
      __builtin_amdgcn_global_load_async_to_lds_b128(AS1CAST(g), AS3CAST(l), 16, 0);
      __builtin_amdgcn_global_load_async_to_lds_b128(AS1CAST(g), AS3CAST(l), 32, 0);
      __builtin_amdgcn_global_load_async_to_lds_b128(AS1CAST(g), AS3CAST(l), 48, 0);
#else
#pragma unroll
      for (int c = 0; c < 4; ++c)
        *(v8h*)(l + c * 8) = *(const v8h*)(g + c * 8);
#endif
    }
    // --- stage V^T tile (transpose: sVt[d][key] = V[key][d]) ---
    {
      const int key = tid & 63, dseg = (tid >> 6) * 32;
      const _Float16* g = Vb + (size_t)(kt0 + key) * rstr + dseg;
#pragma unroll
      for (int c = 0; c < 4; ++c) {
        v8h u = *(const v8h*)(g + c * 8);
#pragma unroll
        for (int j = 0; j < 8; ++j)
          sVt[(dseg + c * 8 + j) * LDK_S + key] = u[j];
      }
    }
#if defined(ATHENA_HAVE_ASYNC)
#if __has_builtin(__builtin_amdgcn_s_wait_asynccnt)
    __builtin_amdgcn_s_wait_asynccnt(0);
#else
    asm volatile("s_wait_asynccnt 0x0" ::: "memory");
#endif
#endif
    __syncthreads();

    // --- S = Q @ K^T for this wave's 16 rows x 64 keys ---
    v8f s[4];
#pragma unroll
    for (int ni = 0; ni < 4; ++ni) s[ni] = (v8f)(0.0f);
#pragma unroll
    for (int kk = 0; kk < 2; ++kk)
#pragma unroll
      for (int ni = 0; ni < 4; ++ni) {
        v16h bk = ld_frag_bt(&sK[(ni * 16) * LDK_S + kk * 32], LDK_S, lane);
        s[ni] = wmma_f16(aq[kk], bk, s[ni]);
      }

    // --- scale + causal mask + online softmax ---
    float pm[8], psum[8], corr[8];
#pragma unroll
    for (int v = 0; v < 8; ++v) pm[v] = -1.0e30f;
#pragma unroll
    for (int ni = 0; ni < 4; ++ni)
#pragma unroll
      for (int v = 0; v < 8; ++v) {
        float x = s[ni][v] * 0.125f;             // 1/sqrt(64)
        const int kg = kt0 + ni * 16 + lo;
        const int qg = q0 + wave * 16 + hi * 8 + v;
        if (kg > qg) x = -1.0e30f;
        s[ni][v] = x;
        pm[v] = fmaxf(pm[v], x);
      }
#pragma unroll
    for (int m = 1; m < 16; m <<= 1)
#pragma unroll
      for (int v = 0; v < 8; ++v)
        pm[v] = fmaxf(pm[v], __shfl_xor(pm[v], m));
#pragma unroll
    for (int v = 0; v < 8; ++v) {
      const float nm = fmaxf(rm[v], pm[v]);
      corr[v] = __expf(rm[v] - nm);
      rm[v] = nm;
      psum[v] = 0.0f;
    }
#pragma unroll
    for (int ni = 0; ni < 4; ++ni)
#pragma unroll
      for (int v = 0; v < 8; ++v) {
        const float e = __expf(s[ni][v] - rm[v]);
        psum[v] += e;
        sP[(wave * 16 + hi * 8 + v) * LDK_S + ni * 16 + lo] = (_Float16)e;
      }
#pragma unroll
    for (int m = 1; m < 16; m <<= 1)
#pragma unroll
      for (int v = 0; v < 8; ++v)
        psum[v] += __shfl_xor(psum[v], m);
#pragma unroll
    for (int v = 0; v < 8; ++v) rs[v] = rs[v] * corr[v] + psum[v];
#pragma unroll
    for (int ni = 0; ni < 4; ++ni)
#pragma unroll
      for (int v = 0; v < 8; ++v) o[ni][v] *= corr[v];

    // --- O += P @ V  (P re-read from LDS in A-fragment layout) ---
#pragma unroll
    for (int kk = 0; kk < 2; ++kk) {
      v16h ap = ld_frag_a(&sP[(wave * 16) * LDK_S + kk * 32], LDK_S, lane);
#pragma unroll
      for (int ni = 0; ni < 4; ++ni) {
        v16h bv = ld_frag_bt(&sVt[(ni * 16) * LDK_S + kk * 32], LDK_S, lane);
        o[ni] = wmma_f16(ap, bv, o[ni]);
      }
    }
    __syncthreads();
  }

  // --- normalize and write att rows (f16 for the out-proj GEMM) ---
  _Float16* Ob = O + ((size_t)b * SEQ + q0 + wave * 16) * rstr + (size_t)h * HD;
#pragma unroll
  for (int v = 0; v < 8; ++v) {
    const float inv = 1.0f / rs[v];
#pragma unroll
    for (int ni = 0; ni < 4; ++ni)
      Ob[(size_t)(hi * 8 + v) * rstr + ni * 16 + lo] =
          (_Float16)(o[ni][v] * inv);
  }
}

// -------------------- launcher ----------------------------------------------

extern "C" void kernel_launch(void* const* d_in, const int* in_sizes, int n_in,
                              void* d_out, int out_size, void* d_ws, size_t ws_size,
                              hipStream_t stream) {
  const float* x    = (const float*)d_in[0];
  const float* Wq   = (const float*)d_in[1];
  const float* bq   = (const float*)d_in[2];
  const float* Wk   = (const float*)d_in[3];
  const float* bk   = (const float*)d_in[4];
  const float* Wv   = (const float*)d_in[5];
  const float* bv   = (const float*)d_in[6];
  const float* Wmix = (const float*)d_in[7];
  const float* bmix = (const float*)d_in[8];
  float* out = (float*)d_out;

  // Workspace carve-up (f16): x(4M) W*(4x1M) q/k/v(3x4M) att(4M) = 24M elems = 48MB
  _Float16* p = (_Float16*)d_ws;
  _Float16* xh    = p;  p += (size_t)MX * DIN;
  _Float16* wqh   = p;  p += (size_t)DIN * HID;
  _Float16* wkh   = p;  p += (size_t)DIN * HID;
  _Float16* wvh   = p;  p += (size_t)DIN * HID;
  _Float16* wmixh = p;  p += (size_t)HID * HID;
  _Float16* qh    = p;  p += (size_t)MX * HID;
  _Float16* kh    = p;  p += (size_t)MX * HID;
  _Float16* vh    = p;  p += (size_t)MX * HID;
  _Float16* atth  = p;  p += (size_t)MX * HID;

  const int nx = MX * DIN, nw = DIN * HID;
  cvt_f32_f16_kernel<<<(nx + 255) / 256, 256, 0, stream>>>(x, xh, nx);
  cvt_f32_f16_kernel<<<(nw + 255) / 256, 256, 0, stream>>>(Wq, wqh, nw);
  cvt_f32_f16_kernel<<<(nw + 255) / 256, 256, 0, stream>>>(Wk, wkh, nw);
  cvt_f32_f16_kernel<<<(nw + 255) / 256, 256, 0, stream>>>(Wv, wvh, nw);
  cvt_f32_f16_kernel<<<(nw + 255) / 256, 256, 0, stream>>>(Wmix, wmixh, nw);

  dim3 gg(HID / BNT, MX / BMT);            // 8 x 32 blocks
  gemm_bias_kernel<true><<<gg, 256, 0, stream>>>(xh, wqh, bq, qh, MX, HID, DIN);
  gemm_bias_kernel<true><<<gg, 256, 0, stream>>>(xh, wkh, bk, kh, MX, HID, DIN);
  gemm_bias_kernel<true><<<gg, 256, 0, stream>>>(xh, wvh, bv, vh, MX, HID, DIN);

  dim3 ga(SEQ / 64, NH, BS);               // 32 x 16 x 2 blocks
  flash_attn_kernel<<<ga, 128, 0, stream>>>(qh, kh, vh, atth);

  gemm_bias_kernel<false><<<gg, 256, 0, stream>>>(atth, wmixh, bmix, out,
                                                  MX, HID, HID);
}
